// PeepholeLSTM_37744172597250
// MI455X (gfx1250) — compile-verified, matrix operand order checked by reading, not run on hardware
//
#include <hip/hip_runtime.h>

// ---------------------------------------------------------------------------
// Peephole LSTM for MI455X (gfx1250, wave32, WMMA bf16 -> f32 accumulate)
// ---------------------------------------------------------------------------

typedef __bf16 bf16;
typedef __attribute__((ext_vector_type(16))) __bf16 v16bf;
typedef __attribute__((ext_vector_type(8)))  __bf16 v8bf;
typedef __attribute__((ext_vector_type(8)))  float  v8f;

#define S_DIM 512
#define B_DIM 64
#define D_DIM 1024
#define H_DIM 1024
#define SB    (S_DIM * B_DIM)              // 32768 rows
#define SBH   ((size_t)SB * H_DIM)         // elements per gate stream

// ---------------------------------------------------------------------------
// Fragment loader for 16-bit A/B matrices (ISA 7.12.2):
//  lane = m + 16*h  (m = row-of-A / col-of-B, h = K-half selector)
//  element e (0..15):  K = k0 + 16*(e/8) + 8*h + (e%8)
//  -> two contiguous 8-element (16B) chunks at +8*h and +16+8*h
// Works for A (rows of X/h, row-major, ld = K-stride) and for B, since
// B(K,N) = W[n,K] with W stored (out,in) row-major.
// ---------------------------------------------------------------------------
__device__ __forceinline__ v16bf load_frag16(const bf16* __restrict__ base,
                                             int row0, int ld, int k0, int lane) {
  const int m = lane & 15;
  const int h = lane >> 4;
  const bf16* p = base + (size_t)(row0 + m) * ld + k0 + 8 * h;
  v8bf lo = *reinterpret_cast<const v8bf*>(p);        // K = k0+8h   .. +7
  v8bf hi = *reinterpret_cast<const v8bf*>(p + 16);   // K = k0+16+8h.. +7
  v16bf r;
#pragma unroll
  for (int i = 0; i < 8; ++i) { r[i] = lo[i]; r[i + 8] = hi[i]; }
  return r;
}

__device__ __forceinline__ float sigmoidf_fast(float x) {
  return 1.0f / (1.0f + __expf(-x));
}

// ---------------------------------------------------------------------------
// Elementwise f32 -> bf16 conversion (grid-stride)
// ---------------------------------------------------------------------------
__global__ __launch_bounds__(256) void cvt_bf16_kernel(const float* __restrict__ src,
                                                       bf16* __restrict__ dst, int n) {
  for (int i = blockIdx.x * blockDim.x + threadIdx.x; i < n;
       i += gridDim.x * blockDim.x)
    dst[i] = (bf16)src[i];
}

__global__ __launch_bounds__(256) void copy_f32_kernel(const float* __restrict__ src,
                                                       float* __restrict__ dst, int n) {
  for (int i = blockIdx.x * blockDim.x + threadIdx.x; i < n;
       i += gridDim.x * blockDim.x)
    dst[i] = src[i];
}

// ---------------------------------------------------------------------------
// Phase 1: gate pre-activations  P[g][srow][h] = x @ W_xg^T + b_g
//   grid = (2048 row-tiles, 32),  block = 256 (8 waves)
//   colgate = blockIdx.y*8 + wave in [0,256): gate = colgate>>6, coltile = colgate&63
//   Each wave: one 16x16 output tile, K = 1024 -> 32 bf16 WMMAs.
// ---------------------------------------------------------------------------
__global__ __launch_bounds__(256) void proj_kernel(
    const bf16* __restrict__ xb,        // [SB, D] bf16
    const bf16* __restrict__ Wx,        // [4][H, D] bf16 (gate order i,f,c,o)
    const float* __restrict__ bi, const float* __restrict__ bff,
    const float* __restrict__ bc, const float* __restrict__ bo,
    float* __restrict__ P) {            // [4][SB, H] f32
  const int lane    = threadIdx.x & 31;
  const int wave    = threadIdx.x >> 5;
  const int colgate = blockIdx.y * 8 + wave;
  const int gate    = colgate >> 6;
  const int n0      = (colgate & 63) * 16;
  const int m0      = blockIdx.x * 16;

  const bf16* W = Wx + (size_t)gate * H_DIM * D_DIM;

  v8f acc = {0.f, 0.f, 0.f, 0.f, 0.f, 0.f, 0.f, 0.f};
#pragma unroll 4
  for (int k = 0; k < D_DIM; k += 32) {
    v16bf a = load_frag16(xb, m0, D_DIM, k, lane);
    v16bf b = load_frag16(W,  n0, D_DIM, k, lane);
    acc = __builtin_amdgcn_wmma_f32_16x16x32_bf16(false, a, false, b,
                                                  (short)0, acc, false, false);
  }

  const float* bg = (gate == 0) ? bi : (gate == 1) ? bff : (gate == 2) ? bc : bo;
  const int n  = lane & 15;
  const int hh = lane >> 4;
  const float bv = bg[n0 + n];
  float* Pg = P + (size_t)gate * SBH;
#pragma unroll
  for (int r = 0; r < 8; ++r) {
    int m = m0 + r + 8 * hh;                       // D layout: M = r + 8*(lane>>4)
    Pg[(size_t)m * H_DIM + n0 + n] = acc[r] + bv;
  }
}

// ---------------------------------------------------------------------------
// Phase 2: one fused recurrent step.
//   grid = (64 h-tiles, 4 b-tiles), block = 256 = 8 waves.
//   wave w: gate g = w&3, ksplit = w>>2 (K halves of 512 -> 16 WMMAs each).
//   LDS reduction across ksplits, then 256-thread pointwise gate update.
// ---------------------------------------------------------------------------
__global__ __launch_bounds__(256) void lstm_step_kernel(
    const bf16* __restrict__ hin,       // [B, H] bf16 (previous hidden)
    bf16* __restrict__ hout,            // [B, H] bf16 (next hidden)
    const bf16* __restrict__ Wh,        // [4][H, H] bf16 (gate order i,f,c,o)
    const float* __restrict__ P,        // [4][SB, H] precomputed x-projections
    const float* __restrict__ Wci, const float* __restrict__ Wcf,
    const float* __restrict__ Wco,
    float* __restrict__ cbuf,           // [B, H] f32, updated in place
    float* __restrict__ out,            // [S, B, H] f32 (d_out)
    int s) {
  __shared__ float tileLds[8][256];

  const int lane = threadIdx.x & 31;
  const int wave = threadIdx.x >> 5;
  const int gate = wave & 3;
  const int ksp  = wave >> 2;
  const int h0   = blockIdx.x * 16;
  const int b0   = blockIdx.y * 16;

  const bf16* W = Wh + (size_t)gate * H_DIM * H_DIM;

  v8f acc = {0.f, 0.f, 0.f, 0.f, 0.f, 0.f, 0.f, 0.f};
  const int kbeg = ksp * (H_DIM / 2);
#pragma unroll 4
  for (int k = kbeg; k < kbeg + H_DIM / 2; k += 32) {
    v16bf a = load_frag16(hin, b0, H_DIM, k, lane);
    v16bf b = load_frag16(W,   h0, H_DIM, k, lane);
    acc = __builtin_amdgcn_wmma_f32_16x16x32_bf16(false, a, false, b,
                                                  (short)0, acc, false, false);
  }

  {
    const int n  = lane & 15;
    const int hh = lane >> 4;
#pragma unroll
    for (int r = 0; r < 8; ++r)
      tileLds[wave][(r + 8 * hh) * 16 + n] = acc[r];
  }
  __syncthreads();

  // Pointwise: one element per thread.
  const int idx = threadIdx.x;          // = M*16 + n
  const int M = idx >> 4, n = idx & 15;
  const int b = b0 + M, hcol = h0 + n;

  const float pre_i = tileLds[0][idx] + tileLds[4][idx];
  const float pre_f = tileLds[1][idx] + tileLds[5][idx];
  const float pre_c = tileLds[2][idx] + tileLds[6][idx];
  const float pre_o = tileLds[3][idx] + tileLds[7][idx];

  const size_t po = ((size_t)s * B_DIM + b) * H_DIM + hcol;
  const size_t bh = (size_t)b * H_DIM + hcol;

  const float xi = P[0 * SBH + po];
  const float xf = P[1 * SBH + po];
  const float xc = P[2 * SBH + po];
  const float xo = P[3 * SBH + po];
  const float cp = cbuf[bh];

  const float i_t  = sigmoidf_fast(xi + pre_i + Wci[hcol] * cp);
  const float f_t  = sigmoidf_fast(xf + pre_f + Wcf[hcol] * cp);
  const float chat = tanhf(xc + pre_c);
  const float cn   = f_t * cp + i_t * chat;
  const float o_t  = sigmoidf_fast(xo + pre_o + Wco[hcol] * cn);
  const float hn   = o_t * tanhf(cn);

  out[po]   = hn;
  cbuf[bh]  = cn;
  hout[bh]  = (bf16)hn;
}

// ---------------------------------------------------------------------------
// Host-side launch sequence (graph-capture safe: only kernel launches and
// async D2D copies on `stream`; all scratch in d_ws).
// ---------------------------------------------------------------------------
extern "C" void kernel_launch(void* const* d_in, const int* in_sizes, int n_in,
                              void* d_out, int out_size, void* d_ws, size_t ws_size,
                              hipStream_t stream) {
  (void)in_sizes; (void)n_in; (void)out_size; (void)ws_size;

  const float* x      = (const float*)d_in[0];
  const float* h_prev = (const float*)d_in[1];
  const float* c_prev = (const float*)d_in[2];
  const float* W_xi = (const float*)d_in[3];
  const float* W_hi = (const float*)d_in[4];
  const float* W_xf = (const float*)d_in[5];
  const float* W_hf = (const float*)d_in[6];
  const float* W_xc = (const float*)d_in[7];
  const float* W_hc = (const float*)d_in[8];
  const float* W_xo = (const float*)d_in[9];
  const float* W_ho = (const float*)d_in[10];
  const float* W_ci = (const float*)d_in[11];
  const float* W_cf = (const float*)d_in[12];
  const float* W_co = (const float*)d_in[13];
  const float* b_i  = (const float*)d_in[14];
  const float* b_f  = (const float*)d_in[15];
  const float* b_c  = (const float*)d_in[16];
  const float* b_o  = (const float*)d_in[17];

  float* out = (float*)d_out;                 // [S,B,H] then h_t, c_t appended

  // ---- workspace layout --------------------------------------------------
  char* ws = (char*)d_ws;
  size_t off = 0;
  bf16* xb  = (bf16*)(ws + off); off += (size_t)SB * D_DIM * 2;        // 64 MB
  bf16* Whb = (bf16*)(ws + off); off += 4ull * H_DIM * H_DIM * 2;      //  8 MB
  bf16* Wxb = (bf16*)(ws + off); off += 4ull * H_DIM * D_DIM * 2;      //  8 MB
  float* P  = (float*)(ws + off); off += 4ull * SBH * 4;               // 512 MB
  bf16* hb0 = (bf16*)(ws + off); off += (size_t)B_DIM * H_DIM * 2;
  bf16* hb1 = (bf16*)(ws + off); off += (size_t)B_DIM * H_DIM * 2;
  float* cb = (float*)(ws + off); off += (size_t)B_DIM * H_DIM * 4;

  // ---- phase 0: precision conversion ------------------------------------
  auto cvt = [&](const float* src, bf16* dst, int n) {
    int blocks = (n + 256 * 8 - 1) / (256 * 8);
    if (blocks < 1) blocks = 1;
    cvt_bf16_kernel<<<blocks, 256, 0, stream>>>(src, dst, n);
  };
  cvt(x, xb, SB * D_DIM);
  cvt(W_hi, Whb + 0ull * H_DIM * H_DIM, H_DIM * H_DIM);
  cvt(W_hf, Whb + 1ull * H_DIM * H_DIM, H_DIM * H_DIM);
  cvt(W_hc, Whb + 2ull * H_DIM * H_DIM, H_DIM * H_DIM);
  cvt(W_ho, Whb + 3ull * H_DIM * H_DIM, H_DIM * H_DIM);
  cvt(W_xi, Wxb + 0ull * H_DIM * D_DIM, H_DIM * D_DIM);
  cvt(W_xf, Wxb + 1ull * H_DIM * D_DIM, H_DIM * D_DIM);
  cvt(W_xc, Wxb + 2ull * H_DIM * D_DIM, H_DIM * D_DIM);
  cvt(W_xo, Wxb + 3ull * H_DIM * D_DIM, H_DIM * D_DIM);
  cvt(h_prev, hb0, B_DIM * H_DIM);
  copy_f32_kernel<<<(B_DIM * H_DIM + 2047) / 2048, 256, 0, stream>>>(
      c_prev, cb, B_DIM * H_DIM);

  // ---- phase 1: batched input projections (big WMMA GEMM) ----------------
  proj_kernel<<<dim3(SB / 16, 32), 256, 0, stream>>>(
      xb, Wxb, b_i, b_f, b_c, b_o, P);

  // ---- phase 2: 512 fused recurrent steps --------------------------------
  bf16* hbuf[2] = {hb0, hb1};
  for (int s = 0; s < S_DIM; ++s) {
    lstm_step_kernel<<<dim3(H_DIM / 16, B_DIM / 16), 256, 0, stream>>>(
        hbuf[s & 1], hbuf[(s + 1) & 1], Whb, P, W_ci, W_cf, W_co, cb, out, s);
  }

  // ---- append h_t (== out[S-1]) and c_t ----------------------------------
  float* h_t_dst = out + SBH;
  float* c_t_dst = out + SBH + (size_t)B_DIM * H_DIM;
  hipMemcpyAsync(h_t_dst, out + (size_t)(S_DIM - 1) * B_DIM * H_DIM,
                 (size_t)B_DIM * H_DIM * sizeof(float),
                 hipMemcpyDeviceToDevice, stream);
  hipMemcpyAsync(c_t_dst, cb, (size_t)B_DIM * H_DIM * sizeof(float),
                 hipMemcpyDeviceToDevice, stream);
}
